// SpeMambaProcessor_49709951484477
// MI455X (gfx1250) — compile-verified, matrix operand order checked by reading, not run on hardware
//
#include <hip/hip_runtime.h>
#include <hip/hip_bf16.h>

typedef __attribute__((ext_vector_type(16))) _Float16 v16h;
typedef __attribute__((ext_vector_type(8)))  float    v8f;

#define D_MODEL 64
#define D_INNER 128
#define D_STATE 16
#define DT_RANK 4
#define L_SEQ   8

// Pre-packed f16 B-fragment store (in d_ws):
//   frags 0..31 : in_proj_w  [256x64]  -> 16 n-tiles x 2 k-frags
//   frags 32..43: x_proj_w   [36x128]  ->  3 n-tiles x 4 k-frags (zero-padded rows)
//   frags 44..59: out_proj_w [64x128]  ->  4 n-tiles x 4 k-frags
// Each fragment: 32 lanes x 16 halves = 512 _Float16 (1 KB), lane-contiguous.
#define FRAG_IN_BASE  0
#define FRAG_XP_BASE  32
#define FRAG_OUT_BASE 44
#define NFRAGS        60

__device__ __forceinline__ float silu_f(float x) { return x / (1.0f + __expf(-x)); }
__device__ __forceinline__ float softplus_f(float x) {
  return (x > 20.0f) ? x : __logf(1.0f + __expf(x));
}

__device__ __forceinline__ v8f wmma16(v16h a, v16h b, v8f c) {
  // D = A(16x32 f16) * B(32x16 f16) + C(16x16 f32)
  return __builtin_amdgcn_wmma_f32_16x16x32_f16(false, a, false, b, (short)0, c,
                                                false, false);
}

// K position inside a 16x32 16-bit A/B fragment for VGPR-pair v (0..7):
// lanes 0-15: K = 0..7 in v0..3, 16..23 in v4..7 ; lanes 16-31: +8 / +8
__device__ __forceinline__ int frag_k(int v, int half) {
  return ((v >> 2) << 4) + (half << 3) + ((v & 3) << 1);
}

// A fragment, elements taken from a single row pointer (per-lane), f32 -> f16
__device__ __forceinline__ v16h afrag_row(const float* __restrict__ row, int k0, int half) {
  v16h f;
#pragma unroll
  for (int v = 0; v < 8; ++v) {
    int k = k0 + frag_k(v, half);
    f[2 * v]     = (_Float16)row[k];
    f[2 * v + 1] = (_Float16)row[k + 1];
  }
  return f;
}

// A fragment from an LDS f32 tile [16][stride]
__device__ __forceinline__ v16h afrag_lds(const float* __restrict__ buf, int stride,
                                          int k0, int lane) {
  int r = lane & 15, half = lane >> 4;
  return afrag_row(buf + r * stride, k0, half);
}

// Pre-packed B fragment: one contiguous 32-byte load per lane.
__device__ __forceinline__ v16h bfrag_pack(const _Float16* __restrict__ pack, int f,
                                           int lane) {
  return *(const v16h*)(pack + (size_t)f * 512 + (size_t)lane * 16);
}

// ---------------------------------------------------------------------------
// Weight pre-pack: f32 weights -> f16 WMMA B-fragment layout in d_ws.
// One wave per fragment; B[k][n] = W[n0+n][k0+k].
// ---------------------------------------------------------------------------
__global__ __launch_bounds__(32) void prepack_kernel(
    const float* __restrict__ in_proj_w,   // [256, 64]
    const float* __restrict__ x_proj_w,    // [36, 128]
    const float* __restrict__ out_proj_w,  // [64, 128]
    _Float16* __restrict__ pack) {
  const int f = blockIdx.x;
  const int lane = threadIdx.x;
  const int half = lane >> 4;

  const float* W;
  int ldw, nmax, n0, k0;
  if (f < FRAG_XP_BASE) {
    W = in_proj_w;  ldw = 64;  nmax = 256;
    n0 = (f >> 1) * 16;              k0 = (f & 1) * 32;
  } else if (f < FRAG_OUT_BASE) {
    int g = f - FRAG_XP_BASE;
    W = x_proj_w;   ldw = 128; nmax = 36;
    n0 = (g >> 2) * 16;              k0 = (g & 3) * 32;
  } else {
    int g = f - FRAG_OUT_BASE;
    W = out_proj_w; ldw = 128; nmax = 64;
    n0 = (g >> 2) * 16;              k0 = (g & 3) * 32;
  }

  const int n = n0 + (lane & 15);
  v16h v;
  if (n < nmax) {
    const float* row = W + (long)n * ldw + k0;
#pragma unroll
    for (int p = 0; p < 8; ++p) {
      int k = frag_k(p, half);
      v[2 * p]     = (_Float16)row[k];
      v[2 * p + 1] = (_Float16)row[k + 1];
    }
  } else {
#pragma unroll
    for (int j = 0; j < 16; ++j) v[j] = (_Float16)0.0f;
  }
  *(v16h*)(pack + (size_t)f * 512 + (size_t)lane * 16) = v;
}

// ---------------------------------------------------------------------------
// Fused bidirectional Mamba: one wave per block, 2 sequences (16 WMMA rows).
// ---------------------------------------------------------------------------
__global__ __launch_bounds__(32) void mamba_bidir_kernel(
    const float* __restrict__ x,          // [nseq, 8, 64] flat
    const _Float16* __restrict__ pack,    // pre-packed weights
    const float* __restrict__ conv_w,     // [128, 4]
    const float* __restrict__ conv_b,     // [128]
    const float* __restrict__ dt_proj_w,  // [128, 4]
    const float* __restrict__ dt_proj_b,  // [128]
    const float* __restrict__ A_log,      // [128, 16]
    const float* __restrict__ Dvec,       // [128]
    float* __restrict__ out,              // [nseq, 8, 64] flat
    int nseq) {
  __shared__ float    s_xin[16 * 128];   // in_proj x-half, original time order
  __shared__ _Float16 s_z[16 * 128];     // gate input, original time order
  __shared__ float    s_xc[16 * 128];    // conv+silu output, scan order (per dir)
  __shared__ float    s_proj[16 * 48];   // x_proj output [dt(4)|B(16)|C(16)|pad]
  __shared__ float    s_yacc[16 * 128];  // fwd+bwd y accumulation, original order
  __shared__ float    s_A[128 * 16];     // A = -exp(A_log)
  __shared__ float    s_cw[128 * 4];
  __shared__ float    s_cb[128];
  __shared__ float    s_dtw[128 * 4];
  __shared__ float    s_dtb[128];
  __shared__ float    s_D[128];

  const int  lane = threadIdx.x;
  const long seqBase = (long)blockIdx.x * 2;
  if (seqBase + 1 >= (long)nseq) return;

  __builtin_prefetch(pack, 0, 0);

  // ---- stage small parameters ----
  for (int i = lane; i < 128 * 16; i += 32) s_A[i] = -__expf(A_log[i]);
  for (int i = lane; i < 128 * 4; i += 32) { s_cw[i] = conv_w[i]; s_dtw[i] = dt_proj_w[i]; }
  for (int i = lane; i < 128; i += 32) { s_cb[i] = conv_b[i]; s_dtb[i] = dt_proj_b[i]; s_D[i] = Dvec[i]; }
  for (int i = lane; i < 16 * 128; i += 32) s_yacc[i] = 0.0f;
  __syncthreads();

  const int r = lane & 15;       // fragment row / C-tile column
  const int half = lane >> 4;

  // ---- input A fragments (rows = 2 seqs x 8 timesteps, K = D_MODEL = 64) ----
  const float* xrow = x + ((seqBase + (r >> 3)) * L_SEQ + (r & 7)) * D_MODEL;
  v16h ax0 = afrag_row(xrow, 0, half);
  v16h ax1 = afrag_row(xrow, 32, half);

  // ---- in_proj: [16x64] x [64x256] -> x_in (e<128) and z (e>=128) ----
  for (int et = 0; et < 16; ++et) {
    v8f c = {};
    c = wmma16(ax0, bfrag_pack(pack, FRAG_IN_BASE + et * 2 + 0, lane), c);
    c = wmma16(ax1, bfrag_pack(pack, FRAG_IN_BASE + et * 2 + 1, lane), c);
#pragma unroll
    for (int v = 0; v < 8; ++v) {
      int m = v + (half << 3);
      float val = c[v];
      if (et < 8) s_xin[m * 128 + et * 16 + r] = val;
      else        s_z[m * 128 + (et - 8) * 16 + r] = (_Float16)val;
    }
  }
  __syncthreads();

  // ---- both directions share x_in / z; out_proj applied once to the sum ----
  for (int dir = 0; dir < 2; ++dir) {
    // causal depthwise conv (+silu) in scan order: fwd t=ls, bwd t=7-ls
    for (int idx = lane; idx < 16 * 128; idx += 32) {
      int m = idx >> 7, ch = idx & 127;
      int sl = m >> 3;   // local sequence (0/1)
      int ls = m & 7;    // scan step
      float acc = s_cb[ch];
#pragma unroll
      for (int k = 0; k < 4; ++k) {
        int lt = dir ? (10 - ls - k) : (ls - 3 + k);  // original-time tap
        float xv = (lt >= 0 && lt < 8) ? s_xin[(sl * 8 + lt) * 128 + ch] : 0.0f;
        acc += s_cw[ch * 4 + k] * xv;
      }
      s_xc[idx] = silu_f(acc);
    }
    __syncthreads();

    // x_proj: [16x128] x [128x36] -> proj (dt_rank | B | C)
    {
      v16h a0 = afrag_lds(s_xc, 128, 0, lane);
      v16h a1 = afrag_lds(s_xc, 128, 32, lane);
      v16h a2 = afrag_lds(s_xc, 128, 64, lane);
      v16h a3 = afrag_lds(s_xc, 128, 96, lane);
      for (int jt = 0; jt < 3; ++jt) {
        v8f c = {};
        c = wmma16(a0, bfrag_pack(pack, FRAG_XP_BASE + jt * 4 + 0, lane), c);
        c = wmma16(a1, bfrag_pack(pack, FRAG_XP_BASE + jt * 4 + 1, lane), c);
        c = wmma16(a2, bfrag_pack(pack, FRAG_XP_BASE + jt * 4 + 2, lane), c);
        c = wmma16(a3, bfrag_pack(pack, FRAG_XP_BASE + jt * 4 + 3, lane), c);
#pragma unroll
        for (int v = 0; v < 8; ++v)
          s_proj[(v + (half << 3)) * 48 + jt * 16 + r] = c[v];
      }
    }
    __syncthreads();

    // selective scan: lane -> (seq = lane>>4, channels ch0+16*i, i<8)
    {
      const int sl = lane >> 4;
      const int ch0 = lane & 15;
      float h[8][16];
#pragma unroll
      for (int i = 0; i < 8; ++i)
#pragma unroll
        for (int s = 0; s < D_STATE; ++s) h[i][s] = 0.0f;

      for (int ls = 0; ls < L_SEQ; ++ls) {
        const int m = sl * 8 + ls;
        const float* pr = s_proj + m * 48;
        const float p0 = pr[0], p1 = pr[1], p2 = pr[2], p3 = pr[3];
        float Bt[D_STATE], Ct[D_STATE];
#pragma unroll
        for (int s = 0; s < D_STATE; ++s) { Bt[s] = pr[4 + s]; Ct[s] = pr[20 + s]; }
        const int torig = dir ? (7 - ls) : ls;
        const int my = sl * 8 + torig;
#pragma unroll
        for (int i = 0; i < 8; ++i) {
          const int ch = ch0 + 16 * i;
          float dt = p0 * s_dtw[ch * 4 + 0] + p1 * s_dtw[ch * 4 + 1] +
                     p2 * s_dtw[ch * 4 + 2] + p3 * s_dtw[ch * 4 + 3] + s_dtb[ch];
          dt = softplus_f(dt);
          const float u = s_xc[m * 128 + ch];
          const float du = dt * u;
          float y = 0.0f;
#pragma unroll
          for (int s = 0; s < D_STATE; ++s) {
            float dA = __expf(dt * s_A[ch * 16 + s]);
            h[i][s] = h[i][s] * dA + du * Bt[s];
            y += h[i][s] * Ct[s];
          }
          y += s_D[ch] * u;
          s_yacc[my * 128 + ch] += y;   // unique (my,ch) per lane, wave-private
        }
      }
    }
    __syncthreads();
  }

  // ---- gate: y *= silu(z) (distributes over fwd+bwd sum) ----
  for (int idx = lane; idx < 16 * 128; idx += 32)
    s_yacc[idx] *= silu_f((float)s_z[idx]);
  __syncthreads();

  // ---- out_proj: [16x128] x [128x64] -> output ----
  {
    v16h y0 = afrag_lds(s_yacc, 128, 0, lane);
    v16h y1 = afrag_lds(s_yacc, 128, 32, lane);
    v16h y2 = afrag_lds(s_yacc, 128, 64, lane);
    v16h y3 = afrag_lds(s_yacc, 128, 96, lane);
    for (int ot = 0; ot < 4; ++ot) {
      v8f c = {};
      c = wmma16(y0, bfrag_pack(pack, FRAG_OUT_BASE + ot * 4 + 0, lane), c);
      c = wmma16(y1, bfrag_pack(pack, FRAG_OUT_BASE + ot * 4 + 1, lane), c);
      c = wmma16(y2, bfrag_pack(pack, FRAG_OUT_BASE + ot * 4 + 2, lane), c);
      c = wmma16(y3, bfrag_pack(pack, FRAG_OUT_BASE + ot * 4 + 3, lane), c);
#pragma unroll
      for (int v = 0; v < 8; ++v) {
        int m = v + (half << 3);
        long nn = seqBase + (m >> 3);
        int ll = m & 7;
        out[(nn * L_SEQ + ll) * D_MODEL + ot * 16 + r] = c[v];
      }
    }
  }
}

extern "C" void kernel_launch(void* const* d_in, const int* in_sizes, int n_in,
                              void* d_out, int out_size, void* d_ws, size_t ws_size,
                              hipStream_t stream) {
  (void)n_in; (void)out_size; (void)ws_size;
  const float* x          = (const float*)d_in[0];
  const float* in_proj_w  = (const float*)d_in[1];
  const float* conv_w     = (const float*)d_in[2];
  const float* conv_b     = (const float*)d_in[3];
  const float* x_proj_w   = (const float*)d_in[4];
  const float* dt_proj_w  = (const float*)d_in[5];
  const float* dt_proj_b  = (const float*)d_in[6];
  const float* A_log      = (const float*)d_in[7];
  const float* Dvec       = (const float*)d_in[8];
  const float* out_proj_w = (const float*)d_in[9];
  float* out = (float*)d_out;
  _Float16* pack = (_Float16*)d_ws;   // 60 frags * 1 KB = 60 KB scratch

  // Stage 1: pack weights into WMMA B-fragment layout (f16).
  prepack_kernel<<<NFRAGS, 32, 0, stream>>>(in_proj_w, x_proj_w, out_proj_w, pack);

  // Stage 2: fused bidirectional Mamba.
  int nseq = in_sizes[0] / (L_SEQ * D_MODEL);   // 16384 for (4,64,64,512)
  int blocks = (nseq + 1) / 2;                  // 2 sequences per 1-wave block
  mamba_bidir_kernel<<<blocks, 32, 0, stream>>>(
      x, pack, conv_w, conv_b, dt_proj_w, dt_proj_b, A_log, Dvec, out, nseq);
}
